// LatticebiLm_76605036691768
// MI455X (gfx1250) — matrix operand order repaired and checked by reading, not compiled, not run on hardware
//
#include <hip/hip_runtime.h>
#include <hip/hip_bf16.h>
#include <math.h>

// ---------------- problem constants (from reference) ----------------
#define T_    256
#define B_    32
#define D_    512
#define H_    512
#define H3_   1536
#define MAXW_ 4
#define L_    2
#define GDIM_ 512
#define RING_ 8          // pending-word ring depth (>= MAXW+1, power of 2)

typedef __attribute__((ext_vector_type(16))) _Float16 v16h;
typedef __attribute__((ext_vector_type(8)))  float    v8f;

union Frag16 {
    v16h v;
    unsigned int u[8];
};

__device__ __forceinline__ float sigm(float x) { return 1.0f / (1.0f + __expf(-x)); }

// ---------------------------------------------------------------
// Convert fp32 weight matrix [K,N] row-major into WMMA-B fragment
// swizzled f16 layout: [N/16][K/32][lane 0..31][16 halves]
//   lane  = (n&15) + 16*(k_local>=16)
//   slot  = k_local & 15      (== vgpr*2 + half)
// ---------------------------------------------------------------
__global__ void swizzle_b_f16(const float* __restrict__ B,
                              _Float16* __restrict__ Bsw, int K, int N) {
    int idx = blockIdx.x * blockDim.x + threadIdx.x;
    if (idx >= K * N) return;
    int k = idx / N, n = idx % N;
    int n_tile = n >> 4;
    int k_tile = k >> 5;
    int k_loc  = k & 31;
    int lane   = (n & 15) + (k_loc & 16);
    int slot   = k_loc & 15;
    size_t dst = (((size_t)n_tile * (K >> 5) + k_tile) * 32 + lane) * 16 + slot;
    Bsw[dst] = (_Float16)B[idx];
}

__global__ void cvt_f32_to_f16(const float* __restrict__ src,
                               _Float16* __restrict__ dst, int n) {
    int i = blockIdx.x * blockDim.x + threadIdx.x;
    if (i < n) dst[i] = (_Float16)src[i];
}

// gather word embeddings -> f16 rows [T*MAXW, GDIM]
__global__ void gather_wemb(const float* __restrict__ gaz,
                            const int* __restrict__ ids,
                            _Float16* __restrict__ out, int total) {
    int i = blockIdx.x * blockDim.x + threadIdx.x;
    if (i >= total) return;
    int row = i / GDIM_, kk = i % GDIM_;
    out[i] = (_Float16)gaz[(size_t)ids[row] * GDIM_ + kk];
}

// ---------------------------------------------------------------
// Generic f16xf16 -> f32 WMMA GEMM, MT output tiles per wave in M,
// compile-time K (KK) so all A-subtile strides become immediate
// offsets and the kt loop has a constant trip count.
//   A:   [M,KK] row-major f16
//   Bsw: swizzled f16 (see swizzle_b_f16)
//   C[m,n] = sum_k A*B + (bias?bias[n]:0) + (addend?addend[(m%ar)*N+n]:0)
// Each wave owns a (16*MT)x16 strip: one B fragment feeds MT WMMAs.
// ---------------------------------------------------------------
template <int MT, int KK>
__global__ void wmma_gemm_f16(const _Float16* __restrict__ A,
                              const _Float16* __restrict__ Bsw,
                              float* __restrict__ C,
                              const float* __restrict__ bias,
                              const float* __restrict__ addend,
                              int M, int N, int addend_rows) {
    const int lane = threadIdx.x & 31;
    const int wave = threadIdx.x >> 5;
    const int tn  = N >> 4;
    const int tmS = M / (16 * MT);
    const int strip = blockIdx.x * (blockDim.x >> 5) + wave;
    if (strip >= tmS * tn) return;
    const int sm = strip / tn, tile_n = strip % tn;
    constexpr int ktc = KK >> 5;

    const int hs = lane >> 4;                       // half-select (lanes 16..31)
    const int m0 = sm * (16 * MT) + (lane & 15);
    // one runtime base; all subtile/kt displacements are constants
    const _Float16* Abase = A + (size_t)m0 * KK + hs * 16;
    const v16h* Bp = (const v16h*)Bsw + (size_t)tile_n * ktc * 32 + lane;

    v8f acc[MT];
#pragma unroll
    for (int i = 0; i < MT; ++i) acc[i] = (v8f){};

#pragma unroll 4
    for (int kt = 0; kt < ktc; ++kt) {
        v16h b = Bp[(size_t)kt * 32];
#pragma unroll
        for (int i = 0; i < MT; ++i) {
            // A fragment: contiguous k order, same as Bsw. VGPR r holds K-pair:
            //   K = 16*hs + 2r
            Frag16 a;
            const unsigned int* A32 =
                (const unsigned int*)(Abase + i * (16 * KK) + (kt << 5));
#pragma unroll
            for (int r = 0; r < 4; ++r) a.u[r] = A32[r];
#pragma unroll
            for (int r = 0; r < 4; ++r) a.u[4 + r] = A32[4 + r];
            acc[i] = __builtin_amdgcn_wmma_f32_16x16x32_f16(
                false, a.v, false, b, (short)0, acc[i], false, false);
        }
    }

    // C/D layout: VGPR v -> row = v + 8*hs ; col = lane&15
    const int col = (tile_n << 4) + (lane & 15);
    float bv = bias ? bias[col] : 0.0f;
#pragma unroll
    for (int i = 0; i < MT; ++i) {
        const int row0 = sm * (16 * MT) + i * 16 + hs * 8;
#pragma unroll
        for (int v = 0; v < 8; ++v) {
            int row = row0 + v;
            float outv = acc[i][v] + bv;
            if (addend) outv += addend[(size_t)(row % addend_rows) * N + col];
            C[(size_t)row * N + col] = outv;
        }
    }
}

// ---------------------------------------------------------------
// per-layer init: state <- h0/c0, zero the pending-word ring
// ---------------------------------------------------------------
__global__ void init_layer(const float* __restrict__ h0, const float* __restrict__ c0,
                           float* __restrict__ h, float* __restrict__ c,
                           _Float16* __restrict__ h_h,
                           float* __restrict__ inc32, _Float16* __restrict__ inc16) {
    int i = blockIdx.x * blockDim.x + threadIdx.x;
    const int BH = B_ * H_;
    if (i < BH) {
        float hv = h0[i];
        h[i] = hv; h_h[i] = (_Float16)hv; c[i] = c0[i];
    }
    if (i < RING_ * MAXW_ * BH) {
        inc32[i] = 0.0f; inc16[i] = (_Float16)0.0f;
    }
}

// ---------------------------------------------------------------
// phase 2: exp-normalized lattice merge + char-cell update
// ---------------------------------------------------------------
__global__ void step_merge(const float* __restrict__ gates,      // [B,3H] = xi_t + h@W_hh + b
                           const float* __restrict__ alpha_pre,  // [MAXW*B,H] = xa_t + inc@aW_hh
                           const float* __restrict__ inc32,      // ring [RING][MAXW][B][H]
                           const int* __restrict__ word_valid,   // [T,MAXW]
                           float* __restrict__ h, float* __restrict__ c,
                           _Float16* __restrict__ h_h,
                           float* __restrict__ out_all,          // outputs_all[li] + t*B*H
                           float* __restrict__ cur_out,          // null or cur + t*B*H
                           _Float16* __restrict__ next_in_h,     // cur_h + t*B*H (f16)
                           int t) {
    int idx = blockIdx.x * blockDim.x + threadIdx.x;
    if (idx >= B_ * H_) return;
    const int b_ = idx / H_, k = idx % H_;

    float i_g = gates[(size_t)b_ * H3_ + k];
    float o_g = gates[(size_t)b_ * H3_ + H_ + k];
    float g_g = gates[(size_t)b_ * H3_ + 2 * H_ + k];
    float ig = sigm(i_g), og = sigm(o_g), gg = tanhf(g_g);

    float ei  = __expf(ig);
    float num = ei * gg;
    float den = ei;
    float incsum = 0.0f;
    const size_t ringbase = (size_t)(t & (RING_ - 1)) * MAXW_ * B_ * H_;
#pragma unroll
    for (int w = 0; w < MAXW_; ++w) {
        int ts = t - w - 1;
        float mval = (ts >= 0) ? (float)word_valid[(size_t)ts * MAXW_ + w] : 0.0f;
        incsum += mval;
        float a  = sigm(alpha_pre[(size_t)(w * B_ + b_) * H_ + k]);
        float ea = mval * __expf(a);
        den += ea;
        num += ea * inc32[ringbase + (size_t)w * B_ * H_ + idx];
    }
    float c_word = num / den;
    float c_no   = (1.0f - ig) * c[idx] + ig * gg;
    float c_new  = (incsum > 0.0f) ? c_word : c_no;
    float h_new  = og * tanhf(c_new);

    c[idx] = c_new;
    h[idx] = h_new;
    h_h[idx] = (_Float16)h_new;
    out_all[idx] = h_new;
    next_in_h[idx] = (_Float16)h_new;
    if (cur_out) cur_out[idx] = h_new;
}

// ---------------------------------------------------------------
// phase 4: word cells starting at t, staged into ring at arrival slot
// ---------------------------------------------------------------
__global__ void word_cells(const float* __restrict__ wcom,   // [B,3H] = h_new@Ww_hh + bw
                           const float* __restrict__ wiw_t,  // [MAXW,3H]
                           const float* __restrict__ c,      // c_new [B,H]
                           float* __restrict__ inc32, _Float16* __restrict__ inc16,
                           int t) {
    int idx = blockIdx.x * blockDim.x + threadIdx.x;
    if (idx >= MAXW_ * B_ * H_) return;
    const int w = idx / (B_ * H_);
    const int rem = idx % (B_ * H_);
    const int b_ = rem / H_, k = rem % H_;

    float iw = wiw_t[(size_t)w * H3_ + k]           + wcom[(size_t)b_ * H3_ + k];
    float fw = wiw_t[(size_t)w * H3_ + H_ + k]      + wcom[(size_t)b_ * H3_ + H_ + k];
    float gw = wiw_t[(size_t)w * H3_ + 2 * H_ + k]  + wcom[(size_t)b_ * H3_ + 2 * H_ + k];
    float cw = sigm(fw) * c[rem] + sigm(iw) * tanhf(gw);

    size_t dst = ((size_t)((t + w + 1) & (RING_ - 1)) * MAXW_ + w) * (size_t)(B_ * H_) + rem;
    inc32[dst] = cw;
    inc16[dst] = (_Float16)cw;
}

// ---------------------------------------------------------------
// host launcher
// ---------------------------------------------------------------
extern "C" void kernel_launch(void* const* d_in, const int* in_sizes, int n_in,
                              void* d_out, int out_size, void* d_ws, size_t ws_size,
                              hipStream_t stream) {
    const float* inputs     = (const float*)d_in[0];
    const float* h0         = (const float*)d_in[1];
    const float* c0         = (const float*)d_in[2];
    const float* gaz        = (const float*)d_in[3];
    const float* W_ih       = (const float*)d_in[4];
    const float* W_hh       = (const float*)d_in[5];
    const float* bvec       = (const float*)d_in[6];
    const float* aW_ih      = (const float*)d_in[7];
    const float* aW_hh      = (const float*)d_in[8];
    const float* a_b        = (const float*)d_in[9];
    const float* Ww_ih      = (const float*)d_in[10];
    const float* Ww_hh      = (const float*)d_in[11];
    const float* bw         = (const float*)d_in[12];
    const int*   word_ids   = (const int*)d_in[13];
    const int*   word_valid = (const int*)d_in[14];

    float* out = (float*)d_out;
    // output layout: cur | hidden_h | hidden_c | outputs_all
    const size_t curBase = 0;
    const size_t hhBase  = (size_t)T_ * B_ * H_;
    const size_t hcBase  = hhBase + (size_t)L_ * B_ * H_;
    const size_t oaBase  = hcBase + (size_t)L_ * B_ * H_;

    // ---- workspace carve (256B aligned) ----
    char* p = (char*)d_ws;
    auto carve = [&](size_t bytes) -> void* {
        void* r = (void*)p;
        p += (bytes + 255) & ~(size_t)255;
        return r;
    };
    _Float16* Wih_sw  = (_Float16*)carve((size_t)D_ * H3_ * 2);
    _Float16* Whh_sw  = (_Float16*)carve((size_t)H_ * H3_ * 2);
    _Float16* aWih_sw = (_Float16*)carve((size_t)D_ * H_ * 2);
    _Float16* aWhh_sw = (_Float16*)carve((size_t)H_ * H_ * 2);
    _Float16* Wwih_sw = (_Float16*)carve((size_t)GDIM_ * H3_ * 2);
    _Float16* Wwhh_sw = (_Float16*)carve((size_t)H_ * H3_ * 2);
    _Float16* cur_h   = (_Float16*)carve((size_t)T_ * B_ * D_ * 2);
    _Float16* wembh   = (_Float16*)carve((size_t)T_ * MAXW_ * GDIM_ * 2);
    float*    xi      = (float*)carve((size_t)T_ * B_ * H3_ * 4);
    float*    xa      = (float*)carve((size_t)T_ * B_ * H_ * 4);
    float*    wiw     = (float*)carve((size_t)T_ * MAXW_ * H3_ * 4);
    float*    gates   = (float*)carve((size_t)B_ * H3_ * 4);
    float*    alphap  = (float*)carve((size_t)MAXW_ * B_ * H_ * 4);
    float*    wcom    = (float*)carve((size_t)B_ * H3_ * 4);
    float*    hbuf    = (float*)carve((size_t)B_ * H_ * 4);
    float*    cbuf    = (float*)carve((size_t)B_ * H_ * 4);
    _Float16* h_h     = (_Float16*)carve((size_t)B_ * H_ * 2);
    float*    inc32   = (float*)carve((size_t)RING_ * MAXW_ * B_ * H_ * 4);
    _Float16* inc16   = (_Float16*)carve((size_t)RING_ * MAXW_ * B_ * H_ * 2);

    const int TPB = 256;
    auto blks = [](size_t n, int t) { return (int)((n + t - 1) / t); };
    // blocks of 128 threads = 4 waves; strips = (M/(16*MT)) * (N/16)
    auto gemm_blocks = [](int M, int N, int MT) {
        return ((M / (16 * MT)) * (N >> 4) + 3) / 4;
    };

    // ---- one-time conversions ----
    swizzle_b_f16<<<blks((size_t)D_ * H3_, TPB), TPB, 0, stream>>>(W_ih,  Wih_sw,  D_, H3_);
    swizzle_b_f16<<<blks((size_t)H_ * H3_, TPB), TPB, 0, stream>>>(W_hh,  Whh_sw,  H_, H3_);
    swizzle_b_f16<<<blks((size_t)D_ * H_,  TPB), TPB, 0, stream>>>(aW_ih, aWih_sw, D_, H_);
    swizzle_b_f16<<<blks((size_t)H_ * H_,  TPB), TPB, 0, stream>>>(aW_hh, aWhh_sw, H_, H_);
    swizzle_b_f16<<<blks((size_t)GDIM_ * H3_, TPB), TPB, 0, stream>>>(Ww_ih, Wwih_sw, GDIM_, H3_);
    swizzle_b_f16<<<blks((size_t)H_ * H3_, TPB), TPB, 0, stream>>>(Ww_hh, Wwhh_sw, H_, H3_);

    cvt_f32_to_f16<<<blks((size_t)T_ * B_ * D_, TPB), TPB, 0, stream>>>(
        inputs, cur_h, T_ * B_ * D_);
    gather_wemb<<<blks((size_t)T_ * MAXW_ * GDIM_, TPB), TPB, 0, stream>>>(
        gaz, word_ids, wembh, T_ * MAXW_ * GDIM_);

    // wiw = word_emb @ Ww_ih : [1024,1536,512]
    wmma_gemm_f16<4, GDIM_><<<gemm_blocks(T_ * MAXW_, H3_, 4), 128, 0, stream>>>(
        wembh, Wwih_sw, wiw, nullptr, nullptr, T_ * MAXW_, H3_, 1);

    for (int li = 0; li < L_; ++li) {
        // xi = cur @ W_ih : [8192,1536,512]   (bias b folded into gates GEMM)
        wmma_gemm_f16<4, D_><<<gemm_blocks(T_ * B_, H3_, 4), 128, 0, stream>>>(
            cur_h, Wih_sw, xi, nullptr, nullptr, T_ * B_, H3_, 1);
        // xa = cur @ aW_ih + a_b : [8192,512,512]
        wmma_gemm_f16<4, D_><<<gemm_blocks(T_ * B_, H_, 4), 128, 0, stream>>>(
            cur_h, aWih_sw, xa, a_b, nullptr, T_ * B_, H_, 1);

        init_layer<<<blks((size_t)RING_ * MAXW_ * B_ * H_, TPB), TPB, 0, stream>>>(
            h0 + (size_t)li * B_ * H_, c0 + (size_t)li * B_ * H_,
            hbuf, cbuf, h_h, inc32, inc16);

        for (int t = 0; t < T_; ++t) {
            // gates = xi_t + h@W_hh + b : [32,1536,512]
            wmma_gemm_f16<2, H_><<<gemm_blocks(B_, H3_, 2), 128, 0, stream>>>(
                h_h, Whh_sw, gates, bvec, xi + (size_t)t * B_ * H3_, B_, H3_, B_);
            // alpha_pre = xa_t + inc_c@aW_hh : [128,512,512]
            wmma_gemm_f16<4, H_><<<gemm_blocks(MAXW_ * B_, H_, 4), 128, 0, stream>>>(
                inc16 + (size_t)(t & (RING_ - 1)) * MAXW_ * B_ * H_, aWhh_sw, alphap,
                nullptr, xa + (size_t)t * B_ * H_, MAXW_ * B_, H_, B_);
            // lattice merge + cell update
            step_merge<<<blks((size_t)B_ * H_, TPB), TPB, 0, stream>>>(
                gates, alphap, inc32, word_valid, hbuf, cbuf, h_h,
                out + oaBase + ((size_t)li * T_ + t) * B_ * H_,
                (li == L_ - 1) ? (out + curBase + (size_t)t * B_ * H_) : nullptr,
                cur_h + (size_t)t * B_ * H_, t);
            // wcommon = h_new@Ww_hh + bw : [32,1536,512]
            wmma_gemm_f16<2, H_><<<gemm_blocks(B_, H3_, 2), 128, 0, stream>>>(
                h_h, Wwhh_sw, wcom, bw, nullptr, B_, H3_, B_);
            // word cells -> ring
            word_cells<<<blks((size_t)MAXW_ * B_ * H_, TPB), TPB, 0, stream>>>(
                wcom, wiw + (size_t)t * MAXW_ * H3_, cbuf, inc32, inc16, t);
        }

        // final states for this layer
        hipMemcpyAsync(out + hhBase + (size_t)li * B_ * H_, hbuf,
                       (size_t)B_ * H_ * sizeof(float), hipMemcpyDeviceToDevice, stream);
        hipMemcpyAsync(out + hcBase + (size_t)li * B_ * H_, cbuf,
                       (size_t)B_ * H_ * sizeof(float), hipMemcpyDeviceToDevice, stream);
    }
}
